// GATLayer_21096879358552
// MI455X (gfx1250) — matrix operand order repaired and checked by reading, not compiled, not run on hardware
//
#include <hip/hip_runtime.h>

typedef __attribute__((ext_vector_type(16))) _Float16 v16h;
typedef __attribute__((ext_vector_type(8)))  float    v8f;
typedef __attribute__((ext_vector_type(4)))  unsigned int u32x4;
typedef __attribute__((ext_vector_type(4)))  int      i32x4;
typedef __attribute__((ext_vector_type(4)))  float    f32x4;

#define B_   4
#define N_   2048
#define F_   256
#define K_   4
#define D_   64
#define CD_  256      // K_*D_
#define MTOT 8192     // B_*N_
#define NEG_SLOPE_ 0.2f

union HV {
    u32x4    u[2];
    v16h     v;
    _Float16 h[16];
};

// ---------------------------------------------------------------- kernel 0a
__global__ void cvt_x(const float* __restrict__ x, _Float16* __restrict__ xh, int n) {
    int i = blockIdx.x * blockDim.x + threadIdx.x;
    if (i < n) xh[i] = (_Float16)x[i];
}

// ---------------------------------------------------------------- kernel 0b
// Pre-swizzle W (K,F,D) into WMMA B-fragment order:
// idx = ((ft*16 + ct)*32 + lane)*16 + h  maps element (f, kk) with
//   f  = ft*32 + (h<8 ? (lane/16)*8 + h : 16 + (lane/16)*8 + h-8)  (K-dim of GEMM, 32 per fragment)
//   kk = ct*16 + lane%16              (output column k*64+d)
__global__ void swz_w(const float* __restrict__ W, _Float16* __restrict__ wf) {
    int i    = blockIdx.x * blockDim.x + threadIdx.x;   // 65536 threads
    int h    = i & 15;
    int lane = (i >> 4) & 31;
    int ct   = (i >> 9) & 15;
    int ft   = (i >> 13) & 7;
    int f    = ft * 32 + (h < 8 ? (lane >> 4) * 8 + h : 16 + (lane >> 4) * 8 + (h - 8));
    int kk   = ct * 16 + (lane & 15);
    int k    = kk >> 6, d = kk & 63;
    wf[i] = (_Float16)W[(k * F_ + f) * D_ + d];
}

// ---------------------------------------------------------------- kernel 1
// Wh_cat (MTOT x 256, f16) = x_f16 (MTOT x 256) @ Wswz (256 x 256)
__global__ void __launch_bounds__(256) gemm_xw(const _Float16* __restrict__ xh,
                                               const _Float16* __restrict__ wf,
                                               _Float16* __restrict__ whc) {
    const int lane  = threadIdx.x & 31;
    const int w     = threadIdx.x >> 5;
    const int lhalf = lane >> 4;
    const int lm    = lane & 15;
    const int m0    = blockIdx.x * 128 + w * 16;
    const int c0t   = blockIdx.y * 4;          // base 16-col tile

    v8f c[4] = {};
    const _Float16* __restrict__ arow = xh + (size_t)(m0 + lm) * F_ + lhalf * 8;
    const _Float16* __restrict__ bbase = wf + (size_t)(c0t * 32 + lane) * 16;

#pragma unroll
    for (int ft = 0; ft < 8; ++ft) {
        HV a;
        a.u[0] = *(const u32x4*)(arow + ft * 32);
        a.u[1] = *(const u32x4*)(arow + ft * 32 + 16);
#pragma unroll
        for (int ns = 0; ns < 4; ++ns) {
            // constant offsets: (ft*16 + ns)*512 halves from per-lane base
            const _Float16* bp = bbase + (size_t)(ft * 16 + ns) * 512;
            HV bb;
            bb.u[0] = ((const u32x4*)bp)[0];
            bb.u[1] = ((const u32x4*)bp)[1];
            c[ns] = __builtin_amdgcn_wmma_f32_16x16x32_f16(
                false, a.v, false, bb.v, (short)0, c[ns], false, false);
        }
    }
#pragma unroll
    for (int ns = 0; ns < 4; ++ns) {
        const int col = (c0t + ns) * 16 + lm;
#pragma unroll
        for (int r = 0; r < 8; ++r) {
            const int row = m0 + lhalf * 8 + r;
            whc[(size_t)row * CD_ + col] = (_Float16)c[ns][r];
        }
    }
}

// ---------------------------------------------------------------- kernel 1b
// e_src/e_dst: per (b,k,n) dot of Wh row (64) with a_src/a_dst (fp32)
__global__ void calc_e(const _Float16* __restrict__ whc,
                       const float* __restrict__ asrc,
                       const float* __restrict__ adst,
                       float* __restrict__ es, float* __restrict__ ed) {
    int i = blockIdx.x * blockDim.x + threadIdx.x;   // 32768 threads
    int n = i & (N_ - 1);
    int k = (i >> 11) & 3;
    int b = i >> 13;
    const _Float16* row = whc + ((size_t)(b * N_ + n)) * CD_ + k * D_;
    float s0 = 0.f, s1 = 0.f;
#pragma unroll 8
    for (int d = 0; d < D_; ++d) {
        float v = (float)row[d];
        s0 += v * asrc[k * D_ + d];
        s1 += v * adst[k * D_ + d];
    }
    es[i] = s0;
    ed[i] = s1;
}

// ---------------------------------------------------------------- kernel 1c
// Swizzle Wh into B-fragment order for the attention GEMM:
// i = ((((b*4+k)*64 + jt)*4 + ns)*32 + lane); element h of that fragment is
//   row = jt*32 + (h<8 ? (lane/16)*8 + h : 16 + (lane/16)*8 + h-8),  col = k*64 + ns*16 + lane%16
__global__ void swz_wh(const _Float16* __restrict__ whc, _Float16* __restrict__ whb) {
    int i    = blockIdx.x * blockDim.x + threadIdx.x;  // 131072 threads
    int lane = i & 31;
    int ns   = (i >> 5) & 3;
    int jt   = (i >> 7) & 63;
    int k    = (i >> 13) & 3;
    int b    = i >> 15;
    const int lhalf = lane >> 4, lm = lane & 15;
    const int col = k * D_ + ns * 16 + lm;
    HV t;
#pragma unroll
    for (int h = 0; h < 16; ++h) {
        const int row = jt * 32 + (h < 8 ? lhalf * 8 + h : 16 + lhalf * 8 + (h - 8));
        t.h[h] = whc[((size_t)(b * N_) + row) * CD_ + col];
    }
    _Float16* dst = whb + (size_t)i * 16;
    ((u32x4*)dst)[0] = t.u[0];
    ((u32x4*)dst)[1] = t.u[1];
}

// ---------------------------------------------------------------- kernel 2
// Flash-style fused GAT attention: one wave = 16 rows, online softmax over
// j-tiles of 32, alpha@Wh via v_wmma_f32_16x16x32_f16.
__global__ void __launch_bounds__(256) gat_attn(const int* __restrict__ adj,
                                                const float* __restrict__ es,
                                                const float* __restrict__ ed,
                                                const _Float16* __restrict__ whb,
                                                float* __restrict__ out) {
    const int lane  = threadIdx.x & 31;
    const int w     = threadIdx.x >> 5;
    const int bid   = blockIdx.x;              // B_*K_*16 = 256 blocks
    const int rb    = bid & 15;
    const int k     = (bid >> 4) & 3;
    const int b     = bid >> 6;
    const int lhalf = lane >> 4, lm = lane & 15;
    const int m0    = rb * 128 + w * 16;
    const int mrow  = m0 + lm;                 // attention row owned by this lane

    const float NEG_INF = -__builtin_inff();
    const float esrc = es[(size_t)(b * K_ + k) * N_ + mrow];

    // per-lane incremented base pointers -> immediate-offset b128 loads
    const int*      __restrict__ aptr = adj + ((size_t)b * N_ + mrow) * N_ + lhalf * 8;
    const float*    __restrict__ eptr = ed + (size_t)(b * K_ + k) * N_ + lhalf * 8;
    const _Float16* __restrict__ bptr = whb + (size_t)(b * K_ + k) * (64 * 4 * 32 * 16)
                                            + (size_t)lane * 16;
    int jbase = lhalf * 8;                     // first j of this lane's low group

    float run_max = NEG_INF, run_sum = 0.f;
    v8f c[4] = {};

    for (int jt = 0; jt < 64; ++jt) {
        i32x4 A0 = *(const i32x4*)(aptr);
        i32x4 A1 = *(const i32x4*)(aptr + 4);
        i32x4 A2 = *(const i32x4*)(aptr + 16);
        i32x4 A3 = *(const i32x4*)(aptr + 20);
        f32x4 E0 = *(const f32x4*)(eptr);
        f32x4 E1 = *(const f32x4*)(eptr + 4);
        f32x4 E2 = *(const f32x4*)(eptr + 16);
        f32x4 E3 = *(const f32x4*)(eptr + 20);

        if (jt < 63) {
            __builtin_prefetch(aptr + 32, 0, 1);   // global_prefetch next adj tile
        }

        int   av[16];
        float ev[16];
#pragma unroll
        for (int q = 0; q < 4; ++q) {
            av[q]      = A0[q];  av[4 + q]  = A1[q];
            av[8 + q]  = A2[q];  av[12 + q] = A3[q];
            ev[q]      = E0[q];  ev[4 + q]  = E1[q];
            ev[8 + q]  = E2[q];  ev[12 + q] = E3[q];
        }

        float eh[16];
        float tmax = NEG_INF;
#pragma unroll
        for (int h = 0; h < 16; ++h) {
            const int j = (h < 8) ? (jbase + h) : (jbase + 16 + (h - 8));
            float e = esrc + ev[h];
            e = (e > 0.f) ? e : NEG_SLOPE_ * e;            // LeakyReLU(0.2)
            const bool msk = (av[h] > 0) || (j == mrow);   // adj | self-loop
            eh[h] = msk ? e : NEG_INF;
            tmax  = fmaxf(tmax, eh[h]);
        }
        tmax = fmaxf(tmax, __shfl_xor(tmax, 16, 32));      // combine row halves

        const float m_new    = fmaxf(run_max, tmax);
        const float meff_new = (m_new == NEG_INF) ? 0.f : m_new;     // isinf(max)->0

        // rescale only when some row's max actually increased (uniform branch);
        // otherwise scale == 1.0 exactly for every lane.
        if (__any(m_new != run_max)) {
            const float meff_old = (run_max == NEG_INF) ? 0.f : run_max;
            const float scale    = __expf(meff_old - meff_new);
            run_sum *= scale;
#pragma unroll
            for (int r = 0; r < 8; ++r) {
                const float sr = __shfl(scale, lhalf * 8 + r, 32);
                c[0][r] *= sr; c[1][r] *= sr; c[2][r] *= sr; c[3][r] *= sr;
            }
            run_max = m_new;
        }

        HV af;
        float lsum = 0.f;
#pragma unroll
        for (int h = 0; h < 16; ++h) {
            const float p = (eh[h] == NEG_INF) ? 0.f : __expf(eh[h] - meff_new);
            lsum += p;
            af.h[h] = (_Float16)p;
        }
        lsum += __shfl_xor(lsum, 16, 32);
        run_sum += lsum;

#pragma unroll
        for (int ns = 0; ns < 4; ++ns) {
            const _Float16* bp = bptr + (size_t)ns * 512;  // constant offsets
            HV bf;
            bf.u[0] = ((const u32x4*)bp)[0];
            bf.u[1] = ((const u32x4*)bp)[1];
            c[ns] = __builtin_amdgcn_wmma_f32_16x16x32_f16(
                false, af.v, false, bf.v, (short)0, c[ns], false, false);
        }

        aptr  += 32;
        eptr  += 32;
        bptr  += 2048;
        jbase += 32;
    }

    // normalize and write out (B, N, K*D) fp32
#pragma unroll
    for (int r = 0; r < 8; ++r) {
        const float denom = __shfl(run_sum, lhalf * 8 + r, 32) + 1e-10f;
        const float inv   = 1.f / denom;
        const int   row   = m0 + lhalf * 8 + r;
#pragma unroll
        for (int ns = 0; ns < 4; ++ns) {
            out[((size_t)(b * N_) + row) * CD_ + k * D_ + ns * 16 + lm] = c[ns][r] * inv;
        }
    }
}

// ---------------------------------------------------------------- launcher
extern "C" void kernel_launch(void* const* d_in, const int* in_sizes, int n_in,
                              void* d_out, int out_size, void* d_ws, size_t ws_size,
                              hipStream_t stream) {
    const float* x     = (const float*)d_in[0];   // (B,N,F)
    const int*   adj   = (const int*)d_in[1];     // (B,N,N)
    const float* W     = (const float*)d_in[2];   // (K,F,D)
    const float* a_src = (const float*)d_in[3];   // (K,D,1)
    const float* a_dst = (const float*)d_in[4];   // (K,D,1)
    float*       out   = (float*)d_out;           // (B,N,K*D)

    char* ws = (char*)d_ws;
    _Float16* xh  = (_Float16*)(ws);                         // 4,194,304 B
    _Float16* wf  = (_Float16*)(ws + 4194304);               //   131,072 B
    _Float16* whc = (_Float16*)(ws + 4325376);               // 4,194,304 B
    _Float16* whb = (_Float16*)(ws + 8519680);               // 4,194,304 B
    float*    es  = (float*)(ws + 12713984);                 //   131,072 B
    float*    ed  = (float*)(ws + 12845056);                 //   131,072 B

    cvt_x<<<8192, 256, 0, stream>>>(x, xh, B_ * N_ * F_);
    swz_w<<<256, 256, 0, stream>>>(W, wf);
    gemm_xw<<<dim3(64, 4), 256, 0, stream>>>(xh, wf, whc);
    calc_e<<<128, 256, 0, stream>>>(whc, a_src, a_dst, es, ed);
    swz_wh<<<512, 256, 0, stream>>>(whc, whb);
    gat_attn<<<256, 256, 0, stream>>>(adj, es, ed, whb, out);
}
